// Attention_7164005450171
// MI455X (gfx1250) — compile-verified, matrix operand order checked by reading, not run on hardware
//
#include <hip/hip_runtime.h>
#include <hip/hip_bf16.h>
#include <math.h>

#define BB 2
#define SS 2048
#define DD 1024
#define HH 16
#define HD 64
#define MM (BB*SS)      // 4096 tokens
#define BH (BB*HH)      // 32 (batch*heads)

typedef __attribute__((ext_vector_type(16))) __bf16 v16bf;
typedef __attribute__((ext_vector_type(8)))  __bf16 v8bf;
typedef __attribute__((ext_vector_type(8)))  float  v8f;
typedef __attribute__((ext_vector_type(4)))  unsigned int u32x4;
typedef __attribute__((ext_vector_type(8)))  int i32x8;
typedef __attribute__((ext_vector_type(4)))  int i32x4;

static __device__ __forceinline__ v16bf cat16(v8bf lo, v8bf hi) {
  return __builtin_shufflevector(lo, hi, 0,1,2,3,4,5,6,7,8,9,10,11,12,13,14,15);
}
static __device__ __forceinline__ v8bf ld8bf(const __bf16* p) {
  return *(const v8bf*)p;   // 16B load (global b128 / ds_load_b128)
}

// ---------------------------------------------------------------------------
// Kernel 0: one-shot f32 -> bf16 conversion (HBM-bound, removes per-tile
// v_cvt traffic from all GEMM inner loops).
// ---------------------------------------------------------------------------
__global__ __launch_bounds__(256)
void cvt_bf16_kernel(const float* __restrict__ src, __bf16* __restrict__ dst,
                     int n8) {
  const int i = blockIdx.x * 256 + threadIdx.x;
  if (i >= n8) return;
  const float4* p = (const float4*)src + (size_t)i * 2;
  float4 a = p[0], b = p[1];
  v8bf r;
  r[0]=(__bf16)a.x; r[1]=(__bf16)a.y; r[2]=(__bf16)a.z; r[3]=(__bf16)a.w;
  r[4]=(__bf16)b.x; r[5]=(__bf16)b.y; r[6]=(__bf16)b.z; r[7]=(__bf16)b.w;
  *(v8bf*)(dst + (size_t)i * 8) = r;
}

// ---------------------------------------------------------------------------
// Kernel 1: fused QKV projection (x @ W^T) + RoPE, bf16 in/out.
// Wave tile 32x64 (2 M x 4 N register blocking): 8 WMMAs per 12 b128 loads.
// grid = (32, 16, 3), block = 128 (4 waves stacked in M -> 128x64 per block)
// ---------------------------------------------------------------------------
__global__ __launch_bounds__(128)
void qkv_rope_kernel(const __bf16* __restrict__ xb,
                     const __bf16* __restrict__ wqb,
                     const __bf16* __restrict__ wkb,
                     const __bf16* __restrict__ wvb,
                     const float* __restrict__ fcos,
                     const float* __restrict__ fsin,
                     __bf16* __restrict__ Qw,
                     __bf16* __restrict__ Kw,
                     __bf16* __restrict__ Vw) {
  const int lane = threadIdx.x & 31;
  const int wave = threadIdx.x >> 5;
  const int half = lane >> 4;
  const int l16  = lane & 15;
  const int z    = blockIdx.z;
  const __bf16* w = (z == 0) ? wqb : (z == 1) ? wkb : wvb;

  const int m0 = (blockIdx.x * 4 + wave) * 32;   // 2 token tiles
  const int n0 = blockIdx.y * 64;                // 4 output-dim tiles

  const __bf16* arow0 = xb + (size_t)(m0 + l16) * DD;
  const __bf16* arow1 = arow0 + (size_t)16 * DD;
  const __bf16* brow[4];
  #pragma unroll
  for (int j = 0; j < 4; ++j) brow[j] = w + (size_t)(n0 + j * 16 + l16) * DD;

  v8f acc[2][4] = {};
  for (int k0 = 0; k0 < DD; k0 += 32) {
    const int ka = k0 + half * 8;   // lane-half K subset per ISA A/B layout
    __builtin_prefetch(arow0 + ka + 256, 0, 3);   // global_prefetch_b8
    __builtin_prefetch(arow1 + ka + 256, 0, 3);
    v16bf a0 = cat16(ld8bf(arow0 + ka), ld8bf(arow0 + ka + 16));
    v16bf a1 = cat16(ld8bf(arow1 + ka), ld8bf(arow1 + ka + 16));
    #pragma unroll
    for (int j = 0; j < 4; ++j) {
      v16bf b = cat16(ld8bf(brow[j] + ka), ld8bf(brow[j] + ka + 16));
      acc[0][j] = __builtin_amdgcn_wmma_f32_16x16x32_bf16(false, a0, false, b,
                                                (short)0, acc[0][j], false, false);
      acc[1][j] = __builtin_amdgcn_wmma_f32_16x16x32_bf16(false, a1, false, b,
                                                (short)0, acc[1][j], false, false);
    }
  }

  const float qscale = (z == 0) ? 0.125f : 1.0f;   // 1/sqrt(HD) folded into Q
  #pragma unroll
  for (int t = 0; t < 2; ++t) {
    #pragma unroll
    for (int j = 0; j < 4; ++j) {
      const int c  = n0 + j * 16 + l16;
      const int h  = c >> 6;
      const int hd = c & 63;
      const int ip = hd >> 1;
      const float sgn = (hd & 1) ? 1.0f : -1.0f;
      #pragma unroll
      for (int i = 0; i < 8; ++i) {
        const int r    = m0 + t * 16 + i + 8 * half;
        const int s    = r & (SS - 1);
        const int bidx = r >> 11;          // r / S
        float v = acc[t][j][i];
        if (z < 2) {                       // RoPE partner lives in lane^1
          float other = __shfl_xor(v, 1, 32);
          v = v * fcos[s * (HD / 2) + ip] + sgn * other * fsin[s * (HD / 2) + ip];
        }
        v *= qscale;
        const size_t off = ((size_t)(bidx * HH + h) * SS + s) * HD + hd;
        const __bf16 bv = (__bf16)v;
        if (z == 0)      Qw[off] = bv;
        else if (z == 1) Kw[off] = bv;
        else             Vw[off] = bv;
      }
    }
  }
}

// ---------------------------------------------------------------------------
// Kernel 2: causal flash attention. block = 128 (4 waves x 32 queries = 128
// queries). K tile staged via the Tensor Data Mover (TDM) where available,
// V tile transposed cooperatively; both shared by all 4 waves.
// ---------------------------------------------------------------------------
__global__ __launch_bounds__(128)
void flash_attn_kernel(const __bf16* __restrict__ Qw,
                       const __bf16* __restrict__ Kw,
                       const __bf16* __restrict__ Vw,
                       __bf16* __restrict__ Aw) {
  __shared__ __align__(16) __bf16 Kl[32 * 64];        // [key][hd]
  __shared__ __align__(16) __bf16 Vt[64 * 32];        // [hd][key] (transposed)
  __shared__ __align__(16) __bf16 Pl[4 * 16 * 32];    // per-wave P slab

  const int tid  = threadIdx.x;
  const int lane = tid & 31;
  const int wave = tid >> 5;
  const int half = lane >> 4;
  const int l16  = lane & 15;

  const int bh  = blockIdx.y;
  const int b   = bh >> 4;
  const int h   = bh & 15;
  const int q0w = blockIdx.x * 128;
  const int q0  = q0w + wave * 32;      // 2 query tiles per wave

  const __bf16* Qbase = Qw + (size_t)bh * SS * HD;
  const __bf16* Kbase = Kw + (size_t)bh * SS * HD;
  const __bf16* Vbase = Vw + (size_t)bh * SS * HD;

  // Q A-operands (2 tiles x two 16x32 K-chunks), loaded once
  v16bf qa[2][2];
  #pragma unroll
  for (int qt = 0; qt < 2; ++qt) {
    const __bf16* qrow = Qbase + (size_t)(q0 + qt * 16 + l16) * HD + half * 8;
    qa[qt][0] = cat16(ld8bf(qrow),      ld8bf(qrow + 16));
    qa[qt][1] = cat16(ld8bf(qrow + 32), ld8bf(qrow + 48));
  }

  v8f o[2][4] = {};
  float rm[2][8], rl[2][8];
  #pragma unroll
  for (int qt = 0; qt < 2; ++qt)
    #pragma unroll
    for (int i = 0; i < 8; ++i) { rm[qt][i] = -1e30f; rl[qt][i] = 0.0f; }

  const int kend = q0w + 128;   // causal: key tiles <= max query in WG
  for (int k0 = 0; k0 < kend; k0 += 32) {
    // ---- K tile staging: TDM DMA (async tensor) with cooperative fallback
#if defined(__gfx1250__) && __has_builtin(__builtin_amdgcn_tensor_load_to_lds)
    if (wave == 0) {
      const unsigned int lds_off = (unsigned int)(uintptr_t)(&Kl[0]);
      const unsigned long long ga =
          (unsigned long long)(uintptr_t)(Kbase + (size_t)k0 * HD);
      u32x4 g0;
      g0[0] = 1u;                               // count=1, user descriptor
      g0[1] = lds_off;                          // lds_addr
      g0[2] = (unsigned int)ga;                 // global_addr[31:0]
      g0[3] = (unsigned int)(ga >> 32) | (2u << 30);  // ga[56:32] | type=2
      i32x8 g1;
      g1[0] = 0x10000;        // data_size = 1 (2 bytes)
      g1[1] = 64 << 16;       // tensor_dim0 = 64
      g1[2] = 2048 << 16;     // tensor_dim1 = 2048
      g1[3] = 64 << 16;       // tile_dim0 = 64
      g1[4] = 32;             // tile_dim1 = 32
      g1[5] = 64;             // tensor_dim0_stride = 64
      g1[6] = 0; g1[7] = 0;
      i32x4 gz = {0, 0, 0, 0};
#if __clang_major__ >= 23
      i32x8 gz8 = {0, 0, 0, 0, 0, 0, 0, 0};
      __builtin_amdgcn_tensor_load_to_lds(g0, g1, gz, gz, gz8, 0);
#else
      __builtin_amdgcn_tensor_load_to_lds(g0, g1, gz, gz, 0);
#endif
      __builtin_amdgcn_s_wait_tensorcnt(0);
    }
#else
    for (int idx = tid * 8; idx < 32 * 64; idx += 128 * 8)
      *(uint4*)&Kl[idx] = *(const uint4*)(Kbase + (size_t)k0 * HD + idx);
#endif
    // ---- V tile transposed staging ----
    for (int idx = tid * 8; idx < 32 * 64; idx += 128 * 8) {
      const int key = idx >> 6;
      const int hd0 = idx & 63;
      uint4 raw = *(const uint4*)(Vbase + (size_t)(k0 + key) * HD + hd0);
      const __bf16* hv = (const __bf16*)&raw;
      #pragma unroll
      for (int j = 0; j < 8; ++j) Vt[(hd0 + j) * 32 + key] = hv[j];
    }
    __syncthreads();

    // K B-operands (shared by both q-tiles)
    const __bf16* kr0 = &Kl[(0  + l16) * 64 + half * 8];
    const __bf16* kr1 = &Kl[(16 + l16) * 64 + half * 8];
    const v16bf b00 = cat16(ld8bf(kr0),      ld8bf(kr0 + 16));
    const v16bf b01 = cat16(ld8bf(kr0 + 32), ld8bf(kr0 + 48));
    const v16bf b10 = cat16(ld8bf(kr1),      ld8bf(kr1 + 16));
    const v16bf b11 = cat16(ld8bf(kr1 + 32), ld8bf(kr1 + 48));
    // V B-operands (shared by both q-tiles)
    v16bf vb[4];
    #pragma unroll
    for (int j = 0; j < 4; ++j) {
      const __bf16* vr = &Vt[(j * 16 + l16) * 32 + half * 8];
      vb[j] = cat16(ld8bf(vr), ld8bf(vr + 16));
    }

    #pragma unroll
    for (int qt = 0; qt < 2; ++qt) {
      v8f s0 = {}, s1 = {};
      s0 = __builtin_amdgcn_wmma_f32_16x16x32_bf16(false, qa[qt][0], false, b00, (short)0, s0, false, false);
      s0 = __builtin_amdgcn_wmma_f32_16x16x32_bf16(false, qa[qt][1], false, b01, (short)0, s0, false, false);
      s1 = __builtin_amdgcn_wmma_f32_16x16x32_bf16(false, qa[qt][0], false, b10, (short)0, s1, false, false);
      s1 = __builtin_amdgcn_wmma_f32_16x16x32_bf16(false, qa[qt][1], false, b11, (short)0, s1, false, false);

      // causal mask + online softmax (row = i + 8*half, col = l16)
      float f[8];
      #pragma unroll
      for (int i = 0; i < 8; ++i) {
        const int mg = q0 + qt * 16 + i + 8 * half;
        float a = (k0 + l16      <= mg) ? s0[i] : -1e30f;
        float c = (k0 + 16 + l16 <= mg) ? s1[i] : -1e30f;
        float mx = fmaxf(a, c);
        #pragma unroll
        for (int d = 8; d >= 1; d >>= 1) mx = fmaxf(mx, __shfl_xor(mx, d, 32));
        const float mnew = fmaxf(rm[qt][i], mx);
        f[i] = __expf(rm[qt][i] - mnew);
        rm[qt][i] = mnew;
        const float p0 = __expf(a - mnew);
        const float p1 = __expf(c - mnew);
        s0[i] = p0; s1[i] = p1;
        float rs = p0 + p1;
        #pragma unroll
        for (int d = 8; d >= 1; d >>= 1) rs += __shfl_xor(rs, d, 32);
        rl[qt][i] = rl[qt][i] * f[i] + rs;
      }
      #pragma unroll
      for (int j = 0; j < 4; ++j)
        #pragma unroll
        for (int i = 0; i < 8; ++i) o[qt][j][i] *= f[i];

      // P: C-layout -> LDS -> A-layout
      __bf16* ps = &Pl[wave * 16 * 32];
      #pragma unroll
      for (int i = 0; i < 8; ++i) {
        const int row = i + 8 * half;
        ps[row * 32 + l16]      = (__bf16)s0[i];
        ps[row * 32 + 16 + l16] = (__bf16)s1[i];
      }
      const __bf16* pr = &ps[l16 * 32 + half * 8];
      const v16bf pa = cat16(ld8bf(pr), ld8bf(pr + 16));

      #pragma unroll
      for (int j = 0; j < 4; ++j)
        o[qt][j] = __builtin_amdgcn_wmma_f32_16x16x32_bf16(false, pa, false, vb[j],
                                                 (short)0, o[qt][j], false, false);
    }
    __syncthreads();
  }

  // finalize: /l, store attn output bf16 [B*S, D] (head-interleaved)
  #pragma unroll
  for (int qt = 0; qt < 2; ++qt) {
    #pragma unroll
    for (int i = 0; i < 8; ++i) {
      const int mg = q0 + qt * 16 + i + 8 * half;
      const float inv = 1.0f / rl[qt][i];
      const size_t row = (size_t)(b * SS + mg) * DD + h * HD;
      #pragma unroll
      for (int j = 0; j < 4; ++j)
        Aw[row + j * 16 + l16] = (__bf16)(o[qt][j][i] * inv);
    }
  }
}

// ---------------------------------------------------------------------------
// Kernel 3: output projection attn(bf16) @ Wo^T(bf16) -> f32.
// Same 32x64 register blocking as kernel 1.
// ---------------------------------------------------------------------------
__global__ __launch_bounds__(128)
void out_proj_kernel(const __bf16* __restrict__ Aw,
                     const __bf16* __restrict__ wob,
                     float* __restrict__ out) {
  const int lane = threadIdx.x & 31;
  const int wave = threadIdx.x >> 5;
  const int half = lane >> 4;
  const int l16  = lane & 15;
  const int m0 = (blockIdx.x * 4 + wave) * 32;
  const int n0 = blockIdx.y * 64;

  const __bf16* arow0 = Aw + (size_t)(m0 + l16) * DD;
  const __bf16* arow1 = arow0 + (size_t)16 * DD;
  const __bf16* brow[4];
  #pragma unroll
  for (int j = 0; j < 4; ++j) brow[j] = wob + (size_t)(n0 + j * 16 + l16) * DD;

  v8f acc[2][4] = {};
  for (int k0 = 0; k0 < DD; k0 += 32) {
    const int ka = k0 + half * 8;
    __builtin_prefetch(arow0 + ka + 256, 0, 3);
    __builtin_prefetch(arow1 + ka + 256, 0, 3);
    v16bf a0 = cat16(ld8bf(arow0 + ka), ld8bf(arow0 + ka + 16));
    v16bf a1 = cat16(ld8bf(arow1 + ka), ld8bf(arow1 + ka + 16));
    #pragma unroll
    for (int j = 0; j < 4; ++j) {
      v16bf b = cat16(ld8bf(brow[j] + ka), ld8bf(brow[j] + ka + 16));
      acc[0][j] = __builtin_amdgcn_wmma_f32_16x16x32_bf16(false, a0, false, b,
                                                (short)0, acc[0][j], false, false);
      acc[1][j] = __builtin_amdgcn_wmma_f32_16x16x32_bf16(false, a1, false, b,
                                                (short)0, acc[1][j], false, false);
    }
  }
  #pragma unroll
  for (int t = 0; t < 2; ++t)
    #pragma unroll
    for (int j = 0; j < 4; ++j)
      #pragma unroll
      for (int i = 0; i < 8; ++i) {
        const int r = m0 + t * 16 + i + 8 * half;
        out[(size_t)r * DD + n0 + j * 16 + l16] = acc[t][j][i];
      }
}

// ---------------------------------------------------------------------------
extern "C" void kernel_launch(void* const* d_in, const int* in_sizes, int n_in,
                              void* d_out, int out_size, void* d_ws, size_t ws_size,
                              hipStream_t stream) {
  const float* x  = (const float*)d_in[0];
  const float* wq = (const float*)d_in[1];
  const float* wk = (const float*)d_in[2];
  const float* wv = (const float*)d_in[3];
  const float* wo = (const float*)d_in[4];
  const float* fc = (const float*)d_in[5];
  const float* fs = (const float*)d_in[6];
  // d_in[7] (dense mask) unused: causal mask applied analytically.

  const size_t nX = (size_t)MM * DD;   // 4M elems
  const size_t nW = (size_t)DD * DD;   // 1M elems
  const size_t nQ = (size_t)BH * SS * HD;  // 4M elems
  __bf16* XA  = (__bf16*)d_ws;         // bf16 x; reused as attn output later
  __bf16* wqb = XA  + nX;
  __bf16* wkb = wqb + nW;
  __bf16* wvb = wkb + nW;
  __bf16* wob = wvb + nW;
  __bf16* Qw  = wob + nW;
  __bf16* Kw  = Qw + nQ;
  __bf16* Vw  = Kw + nQ;
  float* out = (float*)d_out;

  cvt_bf16_kernel<<<dim3((int)(nX / 8 / 256)), 256, 0, stream>>>(x,  XA,  (int)(nX / 8));
  cvt_bf16_kernel<<<dim3((int)(nW / 8 / 256)), 256, 0, stream>>>(wq, wqb, (int)(nW / 8));
  cvt_bf16_kernel<<<dim3((int)(nW / 8 / 256)), 256, 0, stream>>>(wk, wkb, (int)(nW / 8));
  cvt_bf16_kernel<<<dim3((int)(nW / 8 / 256)), 256, 0, stream>>>(wv, wvb, (int)(nW / 8));
  cvt_bf16_kernel<<<dim3((int)(nW / 8 / 256)), 256, 0, stream>>>(wo, wob, (int)(nW / 8));

  qkv_rope_kernel<<<dim3(32, 16, 3), 128, 0, stream>>>(XA, wqb, wkb, wvb, fc, fs,
                                                       Qw, Kw, Vw);
  flash_attn_kernel<<<dim3(SS / 128, BH), 128, 0, stream>>>(Qw, Kw, Vw, XA);
  out_proj_kernel<<<dim3(32, 16), 128, 0, stream>>>(XA, wob, out);
}
